// GAEModel_53764400611652
// MI455X (gfx1250) — matrix operand order repaired and checked by reading, not compiled, not run on hardware
//
#include <hip/hip_runtime.h>
#include <hip/hip_bf16.h>

// ---------------------------------------------------------------------------
// GAE forward: out = sigmoid(Z Z^T), Z from 2 GCN layers on N=8192 nodes.
// bf16 WMMA (v_wmma_f32_16x16x32_bf16) for all three GEMMs. The 8192x8192 f32
// output store (268 MB -> ~11.5us @23.3TB/s) is the roofline floor; each wave
// computes a 16x32 C tile (shared A fragment, 2 accumulators) and the K loop
// is software-pipelined (double-buffered fragments) so WMMAs run with VMEM
// loads still in flight instead of behind s_wait_loadcnt 0.
// Requires K % 64 == 0 (holds: K = 512, 256, 128).
// ---------------------------------------------------------------------------

typedef __attribute__((ext_vector_type(16))) __bf16       v16bf;
typedef __attribute__((ext_vector_type(8)))  float        v8f;
typedef __attribute__((ext_vector_type(8)))  unsigned int v8u;

#define NNODES 8192

__device__ __forceinline__ unsigned short f2bf(float f) {
  unsigned int u = __float_as_uint(f);
  u += 0x7fffu + ((u >> 16) & 1u);   // round-to-nearest-even
  return (unsigned short)(u >> 16);
}

// ---- elementwise helpers ---------------------------------------------------

__global__ void k_cast_bf16(const float* __restrict__ in,
                            unsigned short* __restrict__ out, int n) {
  int i = blockIdx.x * blockDim.x + threadIdx.x;
  if (i < n) out[i] = f2bf(in[i]);
}

// in: K x N row-major f32 ; out: N x K row-major bf16 (transposed B operand)
__global__ void k_transpose_cast(const float* __restrict__ in,
                                 unsigned short* __restrict__ out, int K, int N) {
  int i = blockIdx.x * blockDim.x + threadIdx.x;
  if (i >= K * N) return;
  int k = i / N, n = i % N;
  out[(size_t)n * K + k] = f2bf(in[i]);
}

__global__ void k_fill(float* __restrict__ p, float v, int n) {
  int i = blockIdx.x * blockDim.x + threadIdx.x;
  if (i < n) p[i] = v;
}

__global__ void k_deg_count(const int* __restrict__ dst,
                            float* __restrict__ deg, int E) {
  int e = blockIdx.x * blockDim.x + threadIdx.x;
  if (e < E) atomicAdd(&deg[dst[e]], 1.0f);
}

__global__ void k_rsqrt(float* __restrict__ deg, int n) {
  int i = blockIdx.x * blockDim.x + threadIdx.x;
  if (i < n) deg[i] = rsqrtf(deg[i]);   // deg >= 1 (self loop), no zero case
}

__global__ void k_bias_act_cast(const float* __restrict__ in,
                                const float* __restrict__ bias,
                                unsigned short* __restrict__ out,
                                int total, int F, int relu) {
  int i = blockIdx.x * blockDim.x + threadIdx.x;
  if (i >= total) return;
  float v = in[i] + bias[i % F];
  if (relu) v = fmaxf(v, 0.0f);
  out[i] = f2bf(v);
}

// ---- normalized scatter-add aggregation ------------------------------------
// One wave per edge (incl. synthetic self loops), lanes stride the feature dim.
__global__ void k_aggregate(const float* __restrict__ H,
                            const float* __restrict__ dinv,
                            const int* __restrict__ src,
                            const int* __restrict__ dst,
                            int Eorig, int n, int F, float* __restrict__ out) {
  int wid  = (blockIdx.x * blockDim.x + threadIdx.x) >> 5;
  int lane = threadIdx.x & 31;
  int Etot = Eorig + n;
  if (wid >= Etot) return;
  int s, d;
  if (wid < Eorig) { s = src[wid]; d = dst[wid]; }
  else             { s = wid - Eorig; d = s; }     // self loop
  float c = dinv[s] * dinv[d];
  const float* hrow = H   + (size_t)s * F;
  float*       orow = out + (size_t)d * F;
  for (int f = lane; f < F; f += 32)
    atomicAdd(&orow[f], c * hrow[f]);
}

// ---- bf16 WMMA GEMM: C[MxN] = A[MxK] * Bt[NxK]^T ---------------------------
// One wave owns a 16x32 C tile (one A fragment, two B fragments / accumulators).
// Fragment layouts used here (A and B in one contiguous K order):
//   A (16x32 bf16): lane l<16 -> M=l, VGPR v (0-7): K=2v,2v+1
//                   lane l>=16 -> same M, K offset +16.
//   B (32x16 bf16): lane l%16 = N col, VGPR v: K=2v,2v+1 (+16 for lanes>=16).
//   C/D (16x16 f32): lane l%16 = N col, VGPR r: M = (l/16)*8 + r.
// MODE: 0 = plain f32 store, 1 = sigmoid epilogue.
template <int MODE>
__global__ void gemm_bf16_wmma(const unsigned short* __restrict__ A,
                               const unsigned short* __restrict__ Bt,
                               float* __restrict__ C,
                               int M, int N, int K) {
  int wid = (blockIdx.x * blockDim.x + threadIdx.x) >> 5;
  int tn_count = N >> 5;                 // 32-column tile pairs
  int total = (M >> 4) * tn_count;
  if (wid >= total) return;
  int tm = wid / tn_count;
  int tn = wid % tn_count;
  int lane = threadIdx.x & 31;
  int half = lane >> 4;     // K-half selector
  int l    = lane & 15;     // row (A) / col (B,C)
  int half4 = half * 4;
  int half8 = half * 8;

  const unsigned int* arow =
      (const unsigned int*)(A  + (size_t)(tm * 16 + l) * K);
  const unsigned int* brow0 =
      (const unsigned int*)(Bt + (size_t)(tn * 32 + l) * K);
  const unsigned int* brow1 =
      (const unsigned int*)(Bt + (size_t)(tn * 32 + 16 + l) * K);

  // kk is in u32 (bf16-pair) units; one K-slice of 32 = 16 u32 per row.
#define LOAD_FRAGS(kk, AU, B0, B1)                                        \
  do {                                                                    \
    _Pragma("unroll") for (int v = 0; v < 4; ++v) {                       \
      (AU)[v]     = arow[(kk) + half8 + v];     /* K = 2v   (+half*16) */ \
      (AU)[v + 4] = arow[(kk) + half8 + 4 + v]; /* K = 8+2v (+half*16) */ \
    }                                                                     \
    _Pragma("unroll") for (int v = 0; v < 8; ++v) {                       \
      (B0)[v] = brow0[(kk) + half8 + v];        /* K = 2v (+half*16)   */ \
      (B1)[v] = brow1[(kk) + half8 + v];                                  \
    }                                                                     \
  } while (0)

#define WMMA_PAIR(AU, B0, B1)                                             \
  do {                                                                    \
    v16bf a_  = __builtin_bit_cast(v16bf, (AU));                          \
    v16bf b0_ = __builtin_bit_cast(v16bf, (B0));                          \
    v16bf b1_ = __builtin_bit_cast(v16bf, (B1));                          \
    acc0 = __builtin_amdgcn_wmma_f32_16x16x32_bf16(                       \
        false, a_, false, b0_, (short)0, acc0, false, false);             \
    acc1 = __builtin_amdgcn_wmma_f32_16x16x32_bf16(                       \
        false, a_, false, b1_, (short)0, acc1, false, false);             \
  } while (0)

  v8f acc0 = {}, acc1 = {};
  v8u a0, b00, b10, a1, b01, b11;

  LOAD_FRAGS(0, a0, b00, b10);                 // prologue: stage 0
  for (int k0 = 0; k0 < K; k0 += 64) {         // K % 64 == 0 guaranteed
    int kk = k0 >> 1;
    LOAD_FRAGS(kk + 16, a1, b01, b11);         // stage 1 loads (always valid)
    if (k0 + 96 < K) {                         // near-cache prefetch, 2 ahead
      __builtin_prefetch(arow  + kk + 48, 0, 3);
      __builtin_prefetch(brow0 + kk + 48, 0, 3);
      __builtin_prefetch(brow1 + kk + 48, 0, 3);
    }
    WMMA_PAIR(a0, b00, b10);                   // consume stage 0 under loads
    if (k0 + 64 < K)
      LOAD_FRAGS(kk + 32, a0, b00, b10);       // refill stage 0
    WMMA_PAIR(a1, b01, b11);                   // consume stage 1 under loads
  }
#undef LOAD_FRAGS
#undef WMMA_PAIR

  size_t col = (size_t)tn * 32 + l;
#pragma unroll
  for (int r = 0; r < 8; ++r) {
    int row = tm * 16 + half * 8 + r;
    float* crow = C + (size_t)row * N;
    float v0 = acc0[r];
    float v1 = acc1[r];
    if (MODE == 1) {
      v0 = 1.0f / (1.0f + __expf(-v0));
      v1 = 1.0f / (1.0f + __expf(-v1));
    }
    crow[col]      = v0;
    crow[col + 16] = v1;
  }
}

// ---------------------------------------------------------------------------

extern "C" void kernel_launch(void* const* d_in, const int* in_sizes, int n_in,
                              void* d_out, int out_size, void* d_ws, size_t ws_size,
                              hipStream_t stream) {
  const float* x    = (const float*)d_in[0];   // [N,512]
  const int*   edge = (const int*)  d_in[1];   // [2,E]
  const float* W1   = (const float*)d_in[2];   // [512,256]
  const float* b1   = (const float*)d_in[3];   // [256]
  const float* W2   = (const float*)d_in[4];   // [256,128]
  const float* b2   = (const float*)d_in[5];   // [128]
  float*       out  = (float*)d_out;           // [N,N]

  const int n     = NNODES;
  const int Eorig = in_sizes[1] / 2;
  const int* srcp = edge;
  const int* dstp = edge + Eorig;

  // workspace carve-out (all chunks 256B-aligned)
  char* p = (char*)d_ws;
  auto carve = [&](size_t bytes) {
    void* r = (void*)p;
    p += (bytes + 255) & ~(size_t)255;
    return r;
  };
  unsigned short* Xbf   = (unsigned short*)carve((size_t)n * 512 * 2);
  unsigned short* W1t   = (unsigned short*)carve((size_t)256 * 512 * 2);
  unsigned short* W2t   = (unsigned short*)carve((size_t)128 * 256 * 2);
  float*          deg   = (float*)carve((size_t)n * 4);
  float*          H1    = (float*)carve((size_t)n * 256 * 4);
  float*          H1agg = (float*)carve((size_t)n * 256 * 4);
  unsigned short* H1bf  = (unsigned short*)carve((size_t)n * 256 * 2);
  float*          H2    = (float*)carve((size_t)n * 128 * 4);
  float*          Zagg  = (float*)carve((size_t)n * 128 * 4);
  unsigned short* Zbf   = (unsigned short*)carve((size_t)n * 128 * 2);

  const int T = 256;   // 8 waves per block on wave32
  auto blk = [&](long work) { return (int)((work + T - 1) / T); };

  // bf16 staging of operands (B operands transposed so fragments are contiguous)
  k_cast_bf16     <<<blk((long)n * 512), T, 0, stream>>>(x,  Xbf, n * 512);
  k_transpose_cast<<<blk(512L * 256),    T, 0, stream>>>(W1, W1t, 512, 256);
  k_transpose_cast<<<blk(256L * 128),    T, 0, stream>>>(W2, W2t, 256, 128);

  // symmetric-normalization coefficients: deg = 1 (self loop) + indegree
  k_fill     <<<blk(n),     T, 0, stream>>>(deg, 1.0f, n);
  k_deg_count<<<blk(Eorig), T, 0, stream>>>(dstp, deg, Eorig);
  k_rsqrt    <<<blk(n),     T, 0, stream>>>(deg, n);

  const int Etot = Eorig + n;

  // ---- layer 1: H1 = X*W1 ; H1agg = Â*H1 ; H1bf = bf16(relu(H1agg+b1)) ----
  long tiles1 = (long)(n / 16) * (256 / 32);
  gemm_bf16_wmma<0><<<blk(tiles1 * 32), T, 0, stream>>>(Xbf, W1t, H1, n, 256, 512);
  k_fill       <<<blk((long)n * 256), T, 0, stream>>>(H1agg, 0.0f, n * 256);
  k_aggregate  <<<blk((long)Etot * 32), T, 0, stream>>>(H1, deg, srcp, dstp,
                                                        Eorig, n, 256, H1agg);
  k_bias_act_cast<<<blk((long)n * 256), T, 0, stream>>>(H1agg, b1, H1bf,
                                                        n * 256, 256, 1);

  // ---- layer 2: H2 = H1*W2 ; Zagg = Â*H2 ; Zbf = bf16(Zagg+b2) ------------
  long tiles2 = (long)(n / 16) * (128 / 32);
  gemm_bf16_wmma<0><<<blk(tiles2 * 32), T, 0, stream>>>(H1bf, W2t, H2, n, 128, 256);
  k_fill       <<<blk((long)n * 128), T, 0, stream>>>(Zagg, 0.0f, n * 128);
  k_aggregate  <<<blk((long)Etot * 32), T, 0, stream>>>(H2, deg, srcp, dstp,
                                                        Eorig, n, 128, Zagg);
  k_bias_act_cast<<<blk((long)n * 128), T, 0, stream>>>(Zagg, b2, Zbf,
                                                        n * 128, 128, 0);

  // ---- decoder: out = sigmoid(Z Z^T). Bt == Z itself (row-major N x K) ----
  long tiles3 = (long)(n / 16) * (n / 32);   // 131072 waves
  gemm_bf16_wmma<1><<<blk(tiles3 * 32), T, 0, stream>>>(Zbf, Zbf, out, n, n, 128);
}
